// NonLinearReadoutBlock_19172734009577
// MI455X (gfx1250) — matrix operand order repaired and checked by reading, not compiled, not run on hardware
//
#include <hip/hip_runtime.h>

typedef __attribute__((ext_vector_type(16))) _Float16 v16h;
typedef __attribute__((ext_vector_type(8)))  float    v8f;

#define MUL0 128
#define MUL1 64
#define MUL2 32
#define OUTD 144
#define WAVES 8

// LDS fragment pool offsets (in halves); each frag = 32 lanes * 16 halves = 512
#define W1S_OFF  0
#define W1S_LEN  (14*4*512)
#define W1V1_OFF (W1S_OFF + W1S_LEN)
#define W1V1_LEN (4*2*512)
#define W1V2_OFF (W1V1_OFF + W1V1_LEN)
#define W1V2_LEN (2*1*512)
#define W2S_OFF  (W1V2_OFF + W1V2_LEN)
#define W2S_LEN  (1*4*512)
#define W2V1_OFF (W2S_OFF + W2S_LEN)
#define W2V1_LEN (1*2*512)
#define W2V2_OFF (W2V1_OFF + W2V1_LEN)
#define W2V2_LEN (1*1*512)
#define WTOT     (W2V2_OFF + W2V2_LEN)   // 37376 halves

__device__ __forceinline__ v8f wmma3(v16h ah, v16h al, v16h bh, v16h bl, v8f c) {
  // f16 "hi/lo x3" emulation of an fp32 product, f32 accumulate
  c = __builtin_amdgcn_wmma_f32_16x16x32_f16(false, ah, false, bh, (short)0, c, false, false);
  c = __builtin_amdgcn_wmma_f32_16x16x32_f16(false, ah, false, bl, (short)0, c, false, false);
  c = __builtin_amdgcn_wmma_f32_16x16x32_f16(false, al, false, bh, (short)0, c, false, false);
  return c;
}

// Build one 16x32 A fragment (this lane's slice) from row-major data with
// element stride `estride`. p points at K = (chunk*32 + kg*8) of this lane's row.
// ISA layout: e in 0..7 -> K = kg*8+e ; e in 8..15 -> K = 16+kg*8+(e-8).
__device__ __forceinline__ void make_afrag(const float* p, int estride,
                                           v16h& ah, v16h& al) {
#pragma unroll
  for (int e = 0; e < 8; ++e) {
    float f0 = p[e * estride];
    float f1 = p[(16 + e) * estride];
    _Float16 h0 = (_Float16)f0;
    _Float16 h1 = (_Float16)f1;
    ah[e]     = h0;
    ah[e + 8] = h1;
    al[e]     = (_Float16)(f0 - (float)h0);
    al[e + 8] = (_Float16)(f1 - (float)h1);
  }
}

__device__ __forceinline__ v16h ldfrag(const _Float16* base, int tile, int chunk,
                                       int nchunks, int lane) {
  return *(const v16h*)(base + (((tile * nchunks) + chunk) * 32 + lane) * 16);
}

// Convert a [K x Ncols] row-major fp32 weight into pre-swizzled B-fragments
// (32x16 per tile/chunk), scaled, split into f16 hi/lo.
// B layout: lane n = lane&15, K = chunk*32 + (e>>3)*16 + (lane>>4)*8 + (e&7).
__device__ void fill_frags(const float* __restrict__ W, int Krows, int Ncols,
                           float scale, _Float16* hi, _Float16* lo,
                           int tid, int nthreads) {
  const int nchunks = Krows >> 5;
  const int total = (Ncols >> 4) * nchunks * 512;
  for (int idx = tid; idx < total; idx += nthreads) {
    const int e    = idx & 15;
    const int lane = (idx >> 4) & 31;
    const int rest = idx >> 9;
    const int chunk = rest % nchunks;
    const int tile  = rest / nchunks;
    const int n = tile * 16 + (lane & 15);
    const int k = chunk * 32 + (e >> 3) * 16 + (lane >> 4) * 8 + (e & 7);
    const float f = W[k * Ncols + n] * scale;
    const _Float16 h = (_Float16)f;
    hi[idx] = h;
    lo[idx] = (_Float16)(f - (float)h);
  }
}

__device__ __forceinline__ float sigmoidf_(float x) { return 1.f / (1.f + __expf(-x)); }

__global__ __launch_bounds__(256, 1)
void readout_kernel(const float* __restrict__ x0, const float* __restrict__ x1,
                    const float* __restrict__ x2,
                    const float* __restrict__ W1s, const float* __restrict__ W1v1,
                    const float* __restrict__ W1v2,
                    const float* __restrict__ W2s, const float* __restrict__ W2v1,
                    const float* __restrict__ W2v2,
                    float* __restrict__ out, int ntiles) {
  __shared__ alignas(32) _Float16 g_hi[WTOT];
  __shared__ alignas(32) _Float16 g_lo[WTOT];
  __shared__ alignas(16) float    g_scr[WAVES][16 * MUL0];   // per-wave transpose scratch

  const int tid = threadIdx.x;
  // ---- stage all weights into LDS as pre-swizzled hi/lo B-fragments (once) ----
  fill_frags(W1s,  128, 224, 0.08838834764831845f, g_hi + W1S_OFF,  g_lo + W1S_OFF,  tid, 256);
  fill_frags(W1v1,  64,  64, 0.125f,               g_hi + W1V1_OFF, g_lo + W1V1_OFF, tid, 256);
  fill_frags(W1v2,  32,  32, 0.17677669529663687f, g_hi + W1V2_OFF, g_lo + W1V2_OFF, tid, 256);
  fill_frags(W2s,  128,  16, 0.08838834764831845f, g_hi + W2S_OFF,  g_lo + W2S_OFF,  tid, 256);
  fill_frags(W2v1,  64,  16, 0.125f,               g_hi + W2V1_OFF, g_lo + W2V1_OFF, tid, 256);
  fill_frags(W2v2,  32,  16, 0.17677669529663687f, g_hi + W2V2_OFF, g_lo + W2V2_OFF, tid, 256);
  __syncthreads();

  const int lane = tid & 31;
  const int wave = tid >> 5;
  const int nn   = lane & 15;   // C-tile column / A-tile row for this lane
  const int kg   = lane >> 4;
  const int mrow = kg * 8;      // C-tile row base for this lane
  float* const scr = g_scr[wave];

  for (int tile = blockIdx.x * WAVES + wave; tile < ntiles; tile += gridDim.x * WAVES) {
    const int r0   = tile * 16;
    const int arow = r0 + nn;

    // ---- x0 A-fragments (K = 128 -> 4 chunks), kept in VGPRs ----
    v16h a0h[4], a0l[4];
#pragma unroll
    for (int c = 0; c < 4; ++c)
      make_afrag(x0 + (size_t)arow * MUL0 + c * 32 + kg * 8, 1, a0h[c], a0l[c]);

    // ---- h0 tiles 0..7: silu scalars -> LDS scratch (rolled: low reg pressure) ----
#pragma unroll 2
    for (int T = 0; T < 8; ++T) {
      v8f acc = {0.f,0.f,0.f,0.f,0.f,0.f,0.f,0.f};
#pragma unroll
      for (int c = 0; c < 4; ++c) {
        v16h bh = ldfrag(g_hi + W1S_OFF, T, c, 4, lane);
        v16h bl = ldfrag(g_lo + W1S_OFF, T, c, 4, lane);
        acc = wmma3(a0h[c], a0l[c], bh, bl, acc);
      }
#pragma unroll
      for (int v = 0; v < 8; ++v) {
        float x = acc[v];
        scr[(mrow + v) * MUL0 + T * 16 + nn] = x * sigmoidf_(x);   // silu
      }
    }

    // ---- h0 tiles 8..11: l=1 gates, kept in registers (needs full unroll) ----
    float g1v[4][8];
#pragma unroll
    for (int G = 0; G < 4; ++G) {
      v8f acc = {0.f,0.f,0.f,0.f,0.f,0.f,0.f,0.f};
#pragma unroll
      for (int c = 0; c < 4; ++c) {
        v16h bh = ldfrag(g_hi + W1S_OFF, 8 + G, c, 4, lane);
        v16h bl = ldfrag(g_lo + W1S_OFF, 8 + G, c, 4, lane);
        acc = wmma3(a0h[c], a0l[c], bh, bl, acc);
      }
#pragma unroll
      for (int v = 0; v < 8; ++v) g1v[G][v] = sigmoidf_(acc[v]);
    }

    // ---- h0 tiles 12..13: l=2 gates ----
    float g2v[2][8];
#pragma unroll
    for (int G = 0; G < 2; ++G) {
      v8f acc = {0.f,0.f,0.f,0.f,0.f,0.f,0.f,0.f};
#pragma unroll
      for (int c = 0; c < 4; ++c) {
        v16h bh = ldfrag(g_hi + W1S_OFF, 12 + G, c, 4, lane);
        v16h bl = ldfrag(g_lo + W1S_OFF, 12 + G, c, 4, lane);
        acc = wmma3(a0h[c], a0l[c], bh, bl, acc);
      }
#pragma unroll
      for (int v = 0; v < 8; ++v) g2v[G][v] = sigmoidf_(acc[v]);
    }

    // ---- o0 = silu(s) @ W2s  (A re-read from LDS scratch = transpose) ----
    {
      v8f acc = {0.f,0.f,0.f,0.f,0.f,0.f,0.f,0.f};
#pragma unroll
      for (int c = 0; c < 4; ++c) {
        v16h ah, al;
        make_afrag(scr + nn * MUL0 + c * 32 + kg * 8, 1, ah, al);
        v16h bh = ldfrag(g_hi + W2S_OFF, 0, c, 4, lane);
        v16h bl = ldfrag(g_lo + W2S_OFF, 0, c, 4, lane);
        acc = wmma3(ah, al, bh, bl, acc);
      }
#pragma unroll
      for (int v = 0; v < 8; ++v)
        out[(size_t)(r0 + mrow + v) * OUTD + nn] = acc[v];
    }

    // ---- l=1 vector path: 3 components, K=64 ----
#pragma unroll 1
    for (int i = 0; i < 3; ++i) {
      v16h a1h[2], a1l[2];
#pragma unroll
      for (int c = 0; c < 2; ++c)
        make_afrag(x1 + ((size_t)arow * MUL1 + c * 32 + kg * 8) * 3 + i, 3, a1h[c], a1l[c]);
#pragma unroll
      for (int ut = 0; ut < 4; ++ut) {
        v8f acc = {0.f,0.f,0.f,0.f,0.f,0.f,0.f,0.f};
#pragma unroll
        for (int c = 0; c < 2; ++c) {
          v16h bh = ldfrag(g_hi + W1V1_OFF, ut, c, 2, lane);
          v16h bl = ldfrag(g_lo + W1V1_OFF, ut, c, 2, lane);
          acc = wmma3(a1h[c], a1l[c], bh, bl, acc);
        }
        // gate tile layout == h1 tile layout (same lane -> same (row, w))
#pragma unroll
        for (int v = 0; v < 8; ++v)
          scr[(mrow + v) * MUL1 + ut * 16 + nn] = acc[v] * g1v[ut][v];
      }
      v8f acc = {0.f,0.f,0.f,0.f,0.f,0.f,0.f,0.f};
#pragma unroll
      for (int c = 0; c < 2; ++c) {
        v16h ah, al;
        make_afrag(scr + nn * MUL1 + c * 32 + kg * 8, 1, ah, al);
        v16h bh = ldfrag(g_hi + W2V1_OFF, 0, c, 2, lane);
        v16h bl = ldfrag(g_lo + W2V1_OFF, 0, c, 2, lane);
        acc = wmma3(ah, al, bh, bl, acc);
      }
#pragma unroll
      for (int v = 0; v < 8; ++v)
        out[(size_t)(r0 + mrow + v) * OUTD + 16 + nn * 3 + i] = acc[v];
    }

    // ---- l=2 vector path: 5 components, K=32 ----
#pragma unroll 1
    for (int i = 0; i < 5; ++i) {
      v16h a2h, a2l;
      make_afrag(x2 + ((size_t)arow * MUL2 + kg * 8) * 5 + i, 5, a2h, a2l);
#pragma unroll
      for (int ut = 0; ut < 2; ++ut) {
        v8f acc = {0.f,0.f,0.f,0.f,0.f,0.f,0.f,0.f};
        v16h bh = ldfrag(g_hi + W1V2_OFF, ut, 0, 1, lane);
        v16h bl = ldfrag(g_lo + W1V2_OFF, ut, 0, 1, lane);
        acc = wmma3(a2h, a2l, bh, bl, acc);
#pragma unroll
        for (int v = 0; v < 8; ++v)
          scr[(mrow + v) * MUL2 + ut * 16 + nn] = acc[v] * g2v[ut][v];
      }
      v8f acc = {0.f,0.f,0.f,0.f,0.f,0.f,0.f,0.f};
      {
        v16h ah, al;
        make_afrag(scr + nn * MUL2 + kg * 8, 1, ah, al);
        v16h bh = ldfrag(g_hi + W2V2_OFF, 0, 0, 1, lane);
        v16h bl = ldfrag(g_lo + W2V2_OFF, 0, 0, 1, lane);
        acc = wmma3(ah, al, bh, bl, acc);
      }
#pragma unroll
      for (int v = 0; v < 8; ++v)
        out[(size_t)(r0 + mrow + v) * OUTD + 64 + nn * 5 + i] = acc[v];
    }
  }
}

extern "C" void kernel_launch(void* const* d_in, const int* in_sizes, int n_in,
                              void* d_out, int out_size, void* d_ws, size_t ws_size,
                              hipStream_t stream) {
  (void)n_in; (void)out_size; (void)d_ws; (void)ws_size;
  const float* x0   = (const float*)d_in[0];
  const float* x1   = (const float*)d_in[1];
  const float* x2   = (const float*)d_in[2];
  const float* W1s  = (const float*)d_in[3];
  const float* W1v1 = (const float*)d_in[4];
  const float* W1v2 = (const float*)d_in[5];
  const float* W2s  = (const float*)d_in[6];
  const float* W2v1 = (const float*)d_in[7];
  const float* W2v2 = (const float*)d_in[8];

  const int N = in_sizes[0] / MUL0;     // 200000
  const int ntiles = N / 16;            // 12500 (N divisible by 16)
  int blocks = (ntiles + WAVES - 1) / WAVES;
  if (blocks > 1024) blocks = 1024;
  if (blocks < 1) blocks = 1;

  readout_kernel<<<blocks, 256, 0, stream>>>(x0, x1, x2, W1s, W1v1, W1v2,
                                             W2s, W2v1, W2v2,
                                             (float*)d_out, ntiles);
}
